// Transformer_layer_26663156973900
// MI455X (gfx1250) — matrix operand order repaired and checked by reading, not compiled, not run on hardware
//
#include <hip/hip_runtime.h>

#define DEV __device__ __forceinline__

typedef __attribute__((ext_vector_type(16))) __bf16 bf16x16;
typedef __attribute__((ext_vector_type(8)))  float  f32x8;

union Frag16 { uint4 q[2]; bf16x16 v; };
union Pack8  { unsigned short s[8]; uint4 v; };

DEV unsigned short bf16_of(float f) {
  unsigned int u = __float_as_uint(f);
  return (unsigned short)((u + 0x7FFFu + ((u >> 16) & 1u)) >> 16); // RNE
}

// Low 32 bits of a flat pointer to an LDS object = wave-relative LDS byte addr.
DEV unsigned lds_addr_of(const void* p) {
  return (unsigned)(unsigned long long)p;
}

// CDNA5 async data mover: GLOBAL_LOAD_ASYNC_TO_LDS_B128 (tracked by ASYNCcnt).
DEV void async_copy_b128(unsigned lds_byte_addr, const void* gaddr) {
  asm volatile("global_load_async_to_lds_b128 %0, %1, off"
               :: "v"(lds_byte_addr), "v"(gaddr)
               : "memory");
}

struct GemmArgs {
  long long sAm, sA_bo, sA_bi;   // A row stride, outer/inner batch strides
  long long sBk, sBn, sB_bo, sB_bi;
  long long sCm, sC_bo, sC_bi;
  int nbi;                       // inner batch count (blockIdx.z = bo*nbi+bi)
  float alpha;
};

// C[M,N] (f32) = alpha * A[M,K] (bf16 or f32, converted) * B[K,N] (bf16) + bias
// Tile: 128x64x32, 256 threads (8 waves), each wave: 32x32 via 2x2 WMMA tiles.
// ASYNC=true: both operands bf16 & K-contiguous -> double-buffered
// global_load_async_to_lds_b128 pipeline (loads of tile k+1 overlap WMMA of k).
// ASYNC=false: synchronous staging with on-the-fly f32->bf16 / N->K transpose.
template <typename TA, bool ASYNC>
__global__ __launch_bounds__(256)
void gemm_bf16_wmma(const TA* __restrict__ A, const unsigned short* __restrict__ Bw,
                    float* __restrict__ C, const float* __restrict__ bias,
                    int M, int N, int K, GemmArgs g) {
  constexpr int BM = 128, BN = 64, BK = 32, LDA = 40, LDB = 40; // 80B row pitch
  constexpr int NBUF = ASYNC ? 2 : 1;
  __shared__ unsigned short Al[NBUF][BM * LDA];
  __shared__ unsigned short Bl[NBUF][BN * LDB]; // stored [n][k]

  const int tid  = threadIdx.x;
  const int lane = tid & 31;
  const int wv   = tid >> 5;
  const int wm   = wv & 3;   // 4 M-slabs of 32
  const int wn   = wv >> 2;  // 2 N-slabs of 32
  const int l    = lane & 15;
  const int half = lane >> 4;

  const int tileM = blockIdx.y * BM;
  const int tileN = blockIdx.x * BN;
  const int zo = blockIdx.z / g.nbi;
  const int zi = blockIdx.z % g.nbi;
  const long long offA = (long long)zo * g.sA_bo + (long long)zi * g.sA_bi;
  const long long offB = (long long)zo * g.sB_bo + (long long)zi * g.sB_bi;
  const long long offC = (long long)zo * g.sC_bo + (long long)zi * g.sC_bi;

  // A staging chunk coords: 512 chunks of 8 elems, 2 per thread
  const int am0 = tid >> 2, ak0c = (tid & 3) * 8;          // chunk 0: c = tid
  const int am1 = (tid + 256) >> 2, ak1c = ak0c;           // chunk 1: c = tid+256
  // B weight-layout chunk (contiguous in k): 256 chunks, 1 per thread
  const int bn_w = tid >> 2, bk_w = (tid & 3) * 8;
  // B v-layout chunk (contiguous in n): k = tid>>3, 8 n's starting at (tid&7)*8
  const int bk_v = tid >> 3, bn_v = (tid & 7) * 8;

  const f32x8 zero = {0.f,0.f,0.f,0.f,0.f,0.f,0.f,0.f};
  f32x8 acc[2][2] = {{zero, zero}, {zero, zero}};

  // ---- fragment loads + 2x2 WMMA on one LDS buffer ----
  auto mma_step = [&](const unsigned short* Ab, const unsigned short* Bb) {
    #pragma unroll
    for (int i = 0; i < 2; ++i) {
      const int rowA = wm * 32 + i * 16 + l;
      Frag16 fa;
      // CDNA5 16-bit A(16x32): lane<16: K 0..7 | 16..23 ; lane>=16: 8..15 | 24..31
      fa.q[0] = *(const uint4*)&Ab[rowA * LDA + half * 8];
      fa.q[1] = *(const uint4*)&Ab[rowA * LDA + 16 + half * 8];
      #pragma unroll
      for (int j = 0; j < 2; ++j) {
        const int col = wn * 32 + j * 16 + l;
        Frag16 fb;
        // B(32x16): lane<16: K 0..7 | 16..23 ; lane>=16: 8..15 | 24..31 (n-major staging)
        fb.q[0] = *(const uint4*)&Bb[col * LDB + half * 8];
        fb.q[1] = *(const uint4*)&Bb[col * LDB + 16 + half * 8];
        acc[i][j] = __builtin_amdgcn_wmma_f32_16x16x32_bf16(
            false, fa.v, false, fb.v, (short)0, acc[i][j], false, false);
      }
    }
  };

  if constexpr (ASYNC) {
    // ---- double-buffered async-DMA pipeline (3 B128 async copies/thread) ----
    auto issue = [&](int buf, int k0) {
      async_copy_b128(lds_addr_of(&Al[buf][am0 * LDA + ak0c]),
                      &A[offA + (long long)(tileM + am0) * g.sAm + k0 + ak0c]);
      async_copy_b128(lds_addr_of(&Al[buf][am1 * LDA + ak1c]),
                      &A[offA + (long long)(tileM + am1) * g.sAm + k0 + ak1c]);
      async_copy_b128(lds_addr_of(&Bl[buf][bn_w * LDB + bk_w]),
                      &Bw[offB + (long long)(tileN + bn_w) * g.sBn + k0 + bk_w]);
    };
    issue(0, 0);
    const int nk = K / BK;
    for (int kt = 0; kt < nk; ++kt) {
      const int cur = kt & 1;
      if (kt + 1 < nk) {
        issue(cur ^ 1, (kt + 1) * BK);
        // async loads complete in order: <=3 outstanding => current tile landed
        asm volatile("s_wait_asynccnt 0x3" ::: "memory");
      } else {
        asm volatile("s_wait_asynccnt 0x0" ::: "memory");
      }
      __syncthreads();              // all waves' copies for `cur` visible
      mma_step(Al[cur], Bl[cur]);
      __syncthreads();              // done reading `cur` before it is reissued
    }
  } else {
    // ---- synchronous staging (f32 A conversion / N-contiguous B transpose) ----
    for (int k0 = 0; k0 < K; k0 += BK) {
      if constexpr (sizeof(TA) == 2) {
        const uint4 d0 = *(const uint4*)&A[offA + (long long)(tileM + am0) * g.sAm + k0 + ak0c];
        const uint4 d1 = *(const uint4*)&A[offA + (long long)(tileM + am1) * g.sAm + k0 + ak1c];
        *(uint4*)&Al[0][am0 * LDA + ak0c] = d0;
        *(uint4*)&Al[0][am1 * LDA + ak1c] = d1;
      } else { // f32 -> bf16 on the fly (attention probabilities)
        #pragma unroll
        for (int it = 0; it < 2; ++it) {
          const int m = it ? am1 : am0, kc = it ? ak1c : ak0c;
          const float4 f0 = *(const float4*)&A[offA + (long long)(tileM + m) * g.sAm + k0 + kc];
          const float4 f1 = *(const float4*)&A[offA + (long long)(tileM + m) * g.sAm + k0 + kc + 4];
          Pack8 p;
          p.s[0] = bf16_of(f0.x); p.s[1] = bf16_of(f0.y);
          p.s[2] = bf16_of(f0.z); p.s[3] = bf16_of(f0.w);
          p.s[4] = bf16_of(f1.x); p.s[5] = bf16_of(f1.y);
          p.s[6] = bf16_of(f1.z); p.s[7] = bf16_of(f1.w);
          *(uint4*)&Al[0][m * LDA + kc] = p.v;
        }
      }
      if (g.sBk == 1) { // weight layout: contiguous in k -> pure B128 copy
        const uint4 d = *(const uint4*)&Bw[offB + (long long)(tileN + bn_w) * g.sBn + k0 + bk_w];
        *(uint4*)&Bl[0][bn_w * LDB + bk_w] = d;
      } else {          // v layout (sBn==1): vector load 8 n's, scatter-transpose
        const uint4 d = *(const uint4*)&Bw[offB + (long long)(k0 + bk_v) * g.sBk + tileN + bn_v];
        const unsigned short* s = (const unsigned short*)&d;
        #pragma unroll
        for (int t = 0; t < 8; ++t) Bl[0][(bn_v + t) * LDB + bk_v] = s[t];
      }
      {
        const int kp = (k0 + BK < K) ? (k0 + BK) : k0;
        __builtin_prefetch(&A[offA + (long long)(tileM + am0) * g.sAm + kp + ak0c], 0, 1);
      }
      __syncthreads();
      mma_step(Al[0], Bl[0]);
      __syncthreads();
    }
  }

  // ---- epilogue: C/D layout — VGPR r, lane<16: M=r, lane>=16: M=8+r ----
  #pragma unroll
  for (int i = 0; i < 2; ++i) {
    #pragma unroll
    for (int j = 0; j < 2; ++j) {
      const int col = tileN + wn * 32 + j * 16 + l;
      const float bv = bias ? bias[col] : 0.f;
      #pragma unroll
      for (int r = 0; r < 8; ++r) {
        const int row = tileM + wm * 32 + i * 16 + half * 8 + r;
        C[offC + (long long)row * g.sCm + col] = acc[i][j][r] * g.alpha + bv;
      }
    }
  }
}

// Fused head-mix (W_th) + add attn_prev + stable softmax, IN PLACE on the
// scores tensor living in d_out's attn region. One block per (q, b); all 16
// head rows staged in LDS so mixing reads each score element exactly once.
__global__ __launch_bounds__(256)
void mix_softmax_kernel(float* __restrict__ attn, const float* __restrict__ attn_prev,
                        const float* __restrict__ Wth) {
  const int q = blockIdx.x;
  const int b = blockIdx.y;
  const int tid = threadIdx.x;
  __shared__ float S[16][1024];   // 64 KB of the 320 KB WGP LDS
  __shared__ float red[256];
  __shared__ float wth[256];
  wth[tid] = Wth[tid];            // 16x16 = 256
  for (int h = 0; h < 16; ++h) {
    const float* src = attn + (((long long)(b * 16 + h)) * 1024 + q) * 1024;
    for (int k = tid; k < 1024; k += 256) S[h][k] = src[k];
  }
  __syncthreads();
  for (int g = 0; g < 16; ++g) {
    float vals[4];
    float lmax = -3.4e38f;
    const float* prev = attn_prev + (((long long)(b * 16 + g)) * 1024 + q) * 1024;
    #pragma unroll
    for (int c = 0; c < 4; ++c) {
      const int k = tid + c * 256;
      float a = prev[k];
      #pragma unroll
      for (int h = 0; h < 16; ++h) a += wth[g * 16 + h] * S[h][k];
      vals[c] = a;
      lmax = fmaxf(lmax, a);
    }
    red[tid] = lmax; __syncthreads();
    for (int s = 128; s > 0; s >>= 1) { if (tid < s) red[tid] = fmaxf(red[tid], red[tid + s]); __syncthreads(); }
    const float m = red[0]; __syncthreads();
    float lsum = 0.f;
    #pragma unroll
    for (int c = 0; c < 4; ++c) { vals[c] = __expf(vals[c] - m); lsum += vals[c]; }
    red[tid] = lsum; __syncthreads();
    for (int s = 128; s > 0; s >>= 1) { if (tid < s) red[tid] += red[tid + s]; __syncthreads(); }
    const float inv = 1.0f / red[0]; __syncthreads();
    float* dst = attn + (((long long)(b * 16 + g)) * 1024 + q) * 1024;
    #pragma unroll
    for (int c = 0; c < 4; ++c) dst[tid + c * 256] = vals[c] * inv;
  }
}

// out = LayerNorm(a + b) * gamma + beta ; also emits bf16 copy if outB != null
__global__ __launch_bounds__(256)
void add_layernorm_kernel(const float* __restrict__ A, const float* __restrict__ Bv,
                          const float* __restrict__ gamma, const float* __restrict__ beta,
                          float* __restrict__ outF, unsigned short* __restrict__ outB) {
  const long long row = blockIdx.x;
  const int tid = threadIdx.x;
  __shared__ float red[256];
  float v[4]; float s = 0.f;
  const float* a = A + row * 1024;
  const float* bb = Bv + row * 1024;
  #pragma unroll
  for (int c = 0; c < 4; ++c) { const int j = tid + c * 256; v[c] = a[j] + bb[j]; s += v[c]; }
  red[tid] = s; __syncthreads();
  for (int t = 128; t > 0; t >>= 1) { if (tid < t) red[tid] += red[tid + t]; __syncthreads(); }
  const float mu = red[0] * (1.0f / 1024.0f); __syncthreads();
  float vs = 0.f;
  #pragma unroll
  for (int c = 0; c < 4; ++c) { const float d = v[c] - mu; vs += d * d; }
  red[tid] = vs; __syncthreads();
  for (int t = 128; t > 0; t >>= 1) { if (tid < t) red[tid] += red[tid + t]; __syncthreads(); }
  const float rstd = rsqrtf(red[0] * (1.0f / 1024.0f) + 1e-5f);
  #pragma unroll
  for (int c = 0; c < 4; ++c) {
    const int j = tid + c * 256;
    const float o = (v[c] - mu) * rstd * gamma[j] + beta[j];
    outF[row * 1024 + j] = o;
    if (outB) outB[row * 1024 + j] = bf16_of(o);
  }
}

__global__ __launch_bounds__(256)
void cvt_f32_bf16_kernel(const float* __restrict__ in, unsigned short* __restrict__ out, int n) {
  const int i = blockIdx.x * 256 + threadIdx.x;
  if (i < n) out[i] = bf16_of(in[i]);
}

// ag = h[:, :2048] * relu(h[:, 2048:]) -> bf16
__global__ __launch_bounds__(256)
void gate_kernel(const float* __restrict__ h, unsigned short* __restrict__ out, int n) {
  const int i = blockIdx.x * 256 + threadIdx.x;
  if (i < n) {
    const int row = i >> 11;
    const int col = i & 2047;
    const float a  = h[(long long)row * 4096 + col];
    const float gt = h[(long long)row * 4096 + 2048 + col];
    out[i] = bf16_of(a * fmaxf(gt, 0.f));
  }
}

extern "C" void kernel_launch(void* const* d_in, const int* in_sizes, int n_in,
                              void* d_out, int out_size, void* d_ws, size_t ws_size,
                              hipStream_t stream) {
  const float* x         = (const float*)d_in[0];
  const float* attn_prev = (const float*)d_in[1];
  const float* Wq  = (const float*)d_in[2];
  const float* Wk  = (const float*)d_in[3];
  const float* Wv  = (const float*)d_in[4];
  const float* Wth = (const float*)d_in[5];
  const float* Wpv = (const float*)d_in[6];
  const float* W1  = (const float*)d_in[7];
  const float* b1  = (const float*)d_in[8];
  const float* W2  = (const float*)d_in[9];
  const float* b2  = (const float*)d_in[10];
  const float* ln1_g = (const float*)d_in[11];
  const float* ln1_b = (const float*)d_in[12];
  const float* ln2_g = (const float*)d_in[13];
  const float* ln2_b = (const float*)d_in[14];

  float* outF    = (float*)d_out;                        // (B,P,D) = 4M f32
  float* attnOut = outF + (size_t)4 * 1024 * 1024;       // (B,H,P,P) = 64M f32

  // ---- workspace carve-out ----
  char* ws = (char*)d_ws;
  size_t off = 0;
  auto take = [&](size_t bytes) -> void* {
    void* p = ws + off;
    off = (off + bytes + 255) & ~(size_t)255;
    return p;
  };
  typedef unsigned short u16;
  u16* xb   = (u16*)take((size_t)4194304 * 2);  // x bf16        [4096,1024]
  u16* wqb  = (u16*)take((size_t)1048576 * 2);
  u16* wkb  = (u16*)take((size_t)1048576 * 2);
  u16* wvb  = (u16*)take((size_t)1048576 * 2);
  u16* wpvb = (u16*)take((size_t)1048576 * 2);
  u16* w1b  = (u16*)take((size_t)4194304 * 2);
  u16* w2b  = (u16*)take((size_t)2097152 * 2);
  u16* qb   = (u16*)take((size_t)4194304 * 2);  // q bf16 [4096,1024] (n=h*64+d)
  u16* kb   = (u16*)take((size_t)4194304 * 2);
  u16* vb   = (u16*)take((size_t)4194304 * 2);
  u16* ob   = (u16*)take((size_t)4194304 * 2);  // attn@v output bf16 [4096,1024]
  float* tmpf = (float*)take((size_t)4194304 * 4); // f32 GEMM staging
  float* x1f  = (float*)take((size_t)4194304 * 4);
  u16*   x1b  = (u16*)take((size_t)4194304 * 2);
  float* hf   = (float*)take((size_t)16777216 * 4); // [4096,4096]
  u16*   agb  = (u16*)take((size_t)8388608 * 2);    // [4096,2048]
  float* yf   = (float*)take((size_t)4194304 * 4);
  (void)ws_size; (void)in_sizes; (void)n_in; (void)out_size;

  // ---- 1) f32 -> bf16 conversions ----
  cvt_f32_bf16_kernel<<<4194304 / 256, 256, 0, stream>>>(x,   xb,   4194304);
  cvt_f32_bf16_kernel<<<1048576 / 256, 256, 0, stream>>>(Wq,  wqb,  1048576);
  cvt_f32_bf16_kernel<<<1048576 / 256, 256, 0, stream>>>(Wk,  wkb,  1048576);
  cvt_f32_bf16_kernel<<<1048576 / 256, 256, 0, stream>>>(Wv,  wvb,  1048576);
  cvt_f32_bf16_kernel<<<1048576 / 256, 256, 0, stream>>>(Wpv, wpvb, 1048576);
  cvt_f32_bf16_kernel<<<4194304 / 256, 256, 0, stream>>>(W1,  w1b,  4194304);
  cvt_f32_bf16_kernel<<<2097152 / 256, 256, 0, stream>>>(W2,  w2b,  2097152);

  // ---- 2) QKV projections: [4096,1024] = x @ W^T (f32 out, then -> bf16) ----
  {
    GemmArgs g = {1024, 0, 0,   1, 1024, 0, 0,   1024, 0, 0,   1, 1.0f};
    dim3 grid(1024 / 64, 4096 / 128, 1);
    gemm_bf16_wmma<u16, true><<<grid, 256, 0, stream>>>(xb, wqb, tmpf, nullptr, 4096, 1024, 1024, g);
    cvt_f32_bf16_kernel<<<4194304 / 256, 256, 0, stream>>>(tmpf, qb, 4194304);
    gemm_bf16_wmma<u16, true><<<grid, 256, 0, stream>>>(xb, wkb, tmpf, nullptr, 4096, 1024, 1024, g);
    cvt_f32_bf16_kernel<<<4194304 / 256, 256, 0, stream>>>(tmpf, kb, 4194304);
    gemm_bf16_wmma<u16, true><<<grid, 256, 0, stream>>>(xb, wvb, tmpf, nullptr, 4096, 1024, 1024, g);
    cvt_f32_bf16_kernel<<<4194304 / 256, 256, 0, stream>>>(tmpf, vb, 4194304);
  }

  // ---- 3) scores[b,h,q,k] = (1/8) * q_bh @ k_bh^T, into d_out attn region ----
  {
    GemmArgs g = {1024, 1048576, 64,       // A = q: row p, batch (b, h)
                  1, 1024, 1048576, 64,    // B = k as W[N=pk][K=d]
                  1024, 16777216, 1048576, // C = scores[b,h,:,:]
                  16, 0.125f};
    dim3 grid(1024 / 64, 1024 / 128, 64);
    gemm_bf16_wmma<u16, true><<<grid, 256, 0, stream>>>(qb, kb, attnOut, nullptr, 1024, 1024, 64, g);
  }

  // ---- 4) head mix + attn_prev + softmax (in place) ----
  mix_softmax_kernel<<<dim3(1024, 4), 256, 0, stream>>>(attnOut, attn_prev, Wth);

  // ---- 5) o[b,p,h*64+d] = attn_bh @ v_bh  (A is f32, converted in staging) ----
  {
    GemmArgs g = {1024, 16777216, 1048576, // A = attn
                  1024, 1, 1048576, 64,    // B = v: Bmat[k=pv][n=d]
                  1024, 1048576, 64,       // C = o
                  16, 1.0f};
    dim3 grid(64 / 64, 1024 / 128, 64);
    gemm_bf16_wmma<float, false><<<grid, 256, 0, stream>>>(attnOut, vb, tmpf, nullptr, 1024, 64, 1024, g);
  }
  cvt_f32_bf16_kernel<<<4194304 / 256, 256, 0, stream>>>(tmpf, ob, 4194304);

  // ---- 6) x_sa = o @ Wpv^T ----
  {
    GemmArgs g = {1024, 0, 0,   1, 1024, 0, 0,   1024, 0, 0,   1, 1.0f};
    dim3 grid(1024 / 64, 4096 / 128, 1);
    gemm_bf16_wmma<u16, true><<<grid, 256, 0, stream>>>(ob, wpvb, yf, nullptr, 4096, 1024, 1024, g);
  }

  // ---- 7) x1 = LN(x + x_sa) ----
  add_layernorm_kernel<<<4096, 256, 0, stream>>>(x, yf, ln1_g, ln1_b, x1f, x1b);

  // ---- 8) h = x1 @ W1^T + b1 ----
  {
    GemmArgs g = {1024, 0, 0,   1, 1024, 0, 0,   4096, 0, 0,   1, 1.0f};
    dim3 grid(4096 / 64, 4096 / 128, 1);
    gemm_bf16_wmma<u16, true><<<grid, 256, 0, stream>>>(x1b, w1b, hf, b1, 4096, 4096, 1024, g);
  }

  // ---- 9) ag = a * relu(g) -> bf16 ----
  gate_kernel<<<8388608 / 256, 256, 0, stream>>>(hf, agb, 8388608);

  // ---- 10) y = ag @ W2^T + b2 ----
  {
    GemmArgs g = {2048, 0, 0,   1, 2048, 0, 0,   1024, 0, 0,   1, 1.0f};
    dim3 grid(1024 / 64, 4096 / 128, 1);
    gemm_bf16_wmma<u16, true><<<grid, 256, 0, stream>>>(agb, w2b, yf, b2, 4096, 1024, 2048, g);
  }

  // ---- 11) out = LN(x1 + y) ----
  add_layernorm_kernel<<<4096, 256, 0, stream>>>(x1f, yf, ln2_g, ln2_b, outF, (u16*)nullptr);
}
